// NeuralODE_4647154615581
// MI455X (gfx1250) — matrix operand order repaired and checked by reading, not compiled, not run on hardware
//
#include <hip/hip_runtime.h>
#include <hip/hip_bf16.h>

typedef __attribute__((ext_vector_type(8)))  __bf16 v8bf;
typedef __attribute__((ext_vector_type(16))) __bf16 v16bf;
typedef __attribute__((ext_vector_type(8)))  float  v8f;

#define NWG    128    // persistent workgroups (>= WGP count target, trivially co-resident)
#define BLOCK  128    // 4 waves of 32 (wave32)
// model constants
#define TAU_S  5.0f
#define TAU_M  10.0f
#define TAU_A  20.0f
#define R_GAIN 1.5f
#define KAPPA  0.2f
#define BGIN   0.1f   // W_BG * NU_BG

__device__ __forceinline__ unsigned short f2bf(float f) {
  unsigned u = __float_as_uint(f);
  u += 0x7fffu + ((u >> 16) & 1u);     // round-to-nearest-even
  return (unsigned short)(u >> 16);
}

// device-wide barrier: monotonically increasing arrival counter; target = epoch*NWG
__device__ __forceinline__ void grid_sync(unsigned* bar, unsigned target) {
  __threadfence();
  __syncthreads();
  if (threadIdx.x == 0) {
    __hip_atomic_fetch_add(bar, 1u, __ATOMIC_RELEASE, __HIP_MEMORY_SCOPE_AGENT);
    while (__hip_atomic_load(bar, __ATOMIC_ACQUIRE, __HIP_MEMORY_SCOPE_AGENT) < target)
      __builtin_amdgcn_s_sleep(1);
  }
  __syncthreads();
  __threadfence();
}

// A-fragment: ISA 16-bit A 16x32 layout (lane<16: K 0..7 & 16..23; lane>=16: K 8..15 & 24..31)
__device__ __forceinline__ v16bf load_a(const __bf16* __restrict__ Arow, int k0, int koffA) {
  v8bf x = *(const v8bf*)(Arow + k0 + koffA);
  v8bf y = *(const v8bf*)(Arow + k0 + 16 + koffA);
  return __builtin_shufflevector(x, y, 0,1,2,3,4,5,6,7,8,9,10,11,12,13,14,15);
}
// B-fragment: R chunk broadcast across all 16 columns (lane<16: K 0..7 & 16..23; lane>=16: K 8..15 & 24..31)
__device__ __forceinline__ v16bf load_b(const __bf16* __restrict__ Rb, int k0, int koffB) {
  v8bf x = *(const v8bf*)(Rb + k0 + koffB);
  v8bf y = *(const v8bf*)(Rb + k0 + 16 + koffB);
  return __builtin_shufflevector(x, y, 0,1,2,3,4,5,6,7,8,9,10,11,12,13,14,15);
}

// Two GEMVs over this WG's 16 rows.
// wave w: mat = w&1 selects R-vector, khalf = w>>1 selects K half; each wave runs
// two software-pipelined accumulator chains (quarter-K each) for load/WMMA overlap.
__device__ __forceinline__ void gemv_pair(const __bf16* __restrict__ Wb,
                                          const __bf16* __restrict__ R0,
                                          const __bf16* __restrict__ R1,
                                          float* __restrict__ Mlds,
                                          int N, int wgRow0) {
  const int tid   = threadIdx.x;
  const int wave  = tid >> 5;
  const int lane  = tid & 31;
  const int mat   = wave & 1;
  const int khalf = wave >> 1;                     // 0..1
  const __bf16* __restrict__ Rb = mat ? R1 : R0;
  const int row   = wgRow0 + (lane & 15);
  const int koffA = (lane & 16) ? 8  : 0;
  const int koffB = (lane & 16) ? 8  : 0;
  const __bf16* __restrict__ Arow = Wb + (size_t)row * N;

  const int kb0 = khalf * (N >> 1);                // chain 0 base
  const int kb1 = kb0 + (N >> 2);                  // chain 1 base
  const int NIT = N >> 7;                          // (N/4)/32 iters per chain

  v8f c0 = {}, c1 = {};
  // prologue: stage first fragments of both chains
  v16bf a0 = load_a(Arow, kb0, koffA), b0 = load_b(Rb, kb0, koffB);
  v16bf a1 = load_a(Arow, kb1, koffA), b1 = load_b(Rb, kb1, koffB);
#pragma unroll 2
  for (int i = 1; i < NIT; ++i) {
    const int k0 = kb0 + i * 32, k1 = kb1 + i * 32;
    // issue next fragments before consuming current -> partial loadcnt waits
    v16bf a0n = load_a(Arow, k0, koffA);
    v16bf b0n = load_b(Rb,   k0, koffB);
    v16bf a1n = load_a(Arow, k1, koffA);
    v16bf b1n = load_b(Rb,   k1, koffB);
    __builtin_prefetch(Arow + k0 + 128, 0, 3);
    c0 = __builtin_amdgcn_wmma_f32_16x16x32_bf16(false, a0, false, b0, (short)0, c0,
                                                 false, false);
    c1 = __builtin_amdgcn_wmma_f32_16x16x32_bf16(false, a1, false, b1, (short)0, c1,
                                                 false, false);
    a0 = a0n; b0 = b0n; a1 = a1n; b1 = b1n;
  }
  c0 = __builtin_amdgcn_wmma_f32_16x16x32_bf16(false, a0, false, b0, (short)0, c0,
                                               false, false);
  c1 = __builtin_amdgcn_wmma_f32_16x16x32_bf16(false, a1, false, b1, (short)0, c1,
                                               false, false);
  const v8f cs = c0 + c1;

  // C layout: lane<16 VGPR j = row j (col=lane); lane>=16 VGPR j = row j+8.
  // All columns identical -> lanes 0 and 16 publish; layout [mat][khalf][row]
  if (lane == 0) {
#pragma unroll
    for (int i = 0; i < 8; ++i) Mlds[mat * 32 + khalf * 16 + i] = cs[i];
  } else if (lane == 16) {
#pragma unroll
    for (int i = 0; i < 8; ++i) Mlds[mat * 32 + khalf * 16 + 8 + i] = cs[i];
  }
}

__global__ __launch_bounds__(BLOCK)
void neuralode_rk4_persistent(const float* __restrict__ tg,
                              const float* __restrict__ state,
                              const float* __restrict__ stim,
                              const float* __restrict__ W,
                              float* __restrict__ out,        // [T,4,N]
                              unsigned short* __restrict__ wsW,   // bf16 W, N*N
                              unsigned short* __restrict__ wsR,   // 4 bf16 R vectors
                              unsigned* __restrict__ bar,
                              int N, int T) {
  __shared__ float Mlds[2 * 32];                  // [mat][khalf][16 rows]
  const int tid    = threadIdx.x;
  const int wg     = blockIdx.x;
  const int RPW    = N / NWG;                     // 16 neurons owned per WG
  const int wgRow0 = wg * RPW;
  const int n      = wgRow0 + tid;                // owned neuron (tid < RPW)

  // ---- prologue: quantize W to bf16 (every launch; deterministic) ----
  {
    const int total = N * N;
    for (int i = wg * BLOCK + tid; i < total; i += NWG * BLOCK)
      wsW[i] = f2bf(W[i]);
  }

  // ---- owner-resident state ----
  float yI = 0.f, yH = 0.f, yA = 0.f, yR = 0.f, bconst = 0.f;
  if (tid < RPW) {
    yI = state[0 * N + n];
    yH = state[1 * N + n];
    yA = state[2 * N + n];
    yR = state[3 * N + n];
    bconst = BGIN + stim[n];
    out[(size_t)0 * N + n] = yI;                  // traj[0] = initial state
    out[(size_t)1 * N + n] = yH;
    out[(size_t)2 * N + n] = yA;
    out[(size_t)3 * N + n] = yR;
    const float h2 = 0.5f * (tg[1] - tg[0]);
    wsR[0 * N + n] = f2bf(yR);                    // R1 = y.R
    wsR[1 * N + n] = f2bf(yR + h2 * tanhf(yH - yA));  // R2 (no GEMV needed)
  }

  unsigned epoch = 0;
  grid_sync(bar, ++epoch * NWG);

  const __bf16* Wb  = (const __bf16*)wsW;
  const __bf16* Rb0 = (const __bf16*)(wsR + 0 * N);
  const __bf16* Rb1 = (const __bf16*)(wsR + 1 * N);
  const __bf16* Rb2 = (const __bf16*)(wsR + 2 * N);
  const __bf16* Rb3 = (const __bf16*)(wsR + 3 * N);

  float k1I = 0, k1H = 0, k1A = 0, k1R = 0;
  float k2I = 0, k2H = 0, k2A = 0, k2R = 0;

  for (int s = 0; s < T - 1; ++s) {
    const float dt = tg[s + 1] - tg[s];
    const float h2 = 0.5f * dt;

    // ---- phase A: M1 = W@R1, M2 = W@R2 (independent) ----
    gemv_pair(Wb, Rb0, Rb1, Mlds, N, wgRow0);
    __syncthreads();
    if (tid < RPW) {
      const float M1 = Mlds[tid] + Mlds[16 + tid];
      const float M2 = Mlds[32 + tid] + Mlds[48 + tid];
      k1I = -yI / TAU_S + M1 + bconst;
      k1H = (-yH + R_GAIN * k1I) / TAU_M;
      k1A = (-yA + KAPPA * yR) / TAU_A;
      k1R = tanhf(yH - yA);
      const float I2 = yI + h2 * k1I, H2 = yH + h2 * k1H;
      const float A2 = yA + h2 * k1A, R2 = yR + h2 * k1R;
      k2I = -I2 / TAU_S + M2 + bconst;
      k2H = (-H2 + R_GAIN * k2I) / TAU_M;
      k2A = (-A2 + KAPPA * R2) / TAU_A;
      k2R = tanhf(H2 - A2);
      const float H3 = yH + h2 * k2H, A3 = yA + h2 * k2A, R3 = yR + h2 * k2R;
      const float k3R = tanhf(H3 - A3);
      wsR[2 * N + n] = f2bf(R3);                  // R3
      wsR[3 * N + n] = f2bf(yR + dt * k3R);       // R4 (no GEMV needed)
    }
    grid_sync(bar, ++epoch * NWG);

    // ---- phase B: M3 = W@R3, M4 = W@R4 (independent) ----
    gemv_pair(Wb, Rb2, Rb3, Mlds, N, wgRow0);
    __syncthreads();
    if (tid < RPW) {
      const float M3 = Mlds[tid] + Mlds[16 + tid];
      const float M4 = Mlds[32 + tid] + Mlds[48 + tid];
      const float I3 = yI + h2 * k2I, H3 = yH + h2 * k2H;
      const float A3 = yA + h2 * k2A, R3 = yR + h2 * k2R;
      const float k3I = -I3 / TAU_S + M3 + bconst;
      const float k3H = (-H3 + R_GAIN * k3I) / TAU_M;
      const float k3A = (-A3 + KAPPA * R3) / TAU_A;
      const float k3R = tanhf(H3 - A3);
      const float I4 = yI + dt * k3I, H4 = yH + dt * k3H;
      const float A4 = yA + dt * k3A, R4 = yR + dt * k3R;
      const float k4I = -I4 / TAU_S + M4 + bconst;
      const float k4H = (-H4 + R_GAIN * k4I) / TAU_M;
      const float k4A = (-A4 + KAPPA * R4) / TAU_A;
      const float k4R = tanhf(H4 - A4);
      const float w6 = dt / 6.0f;
      yI += w6 * (k1I + 2.f * k2I + 2.f * k3I + k4I);
      yH += w6 * (k1H + 2.f * k2H + 2.f * k3H + k4H);
      yA += w6 * (k1A + 2.f * k2A + 2.f * k3A + k4A);
      yR += w6 * (k1R + 2.f * k2R + 2.f * k3R + k4R);
      const size_t o = (size_t)(s + 1) * 4 * N;
      out[o + 0 * N + n] = yI;
      out[o + 1 * N + n] = yH;
      out[o + 2 * N + n] = yA;
      out[o + 3 * N + n] = yR;
      if (s + 1 < T - 1) {                        // seed next step's phase A
        const float h2n = 0.5f * (tg[s + 2] - tg[s + 1]);
        wsR[0 * N + n] = f2bf(yR);
        wsR[1 * N + n] = f2bf(yR + h2n * tanhf(yH - yA));
      }
    }
    grid_sync(bar, ++epoch * NWG);
  }
}

extern "C" void kernel_launch(void* const* d_in, const int* in_sizes, int n_in,
                              void* d_out, int out_size, void* d_ws, size_t ws_size,
                              hipStream_t stream) {
  const float* tg    = (const float*)d_in[0];
  const float* state = (const float*)d_in[1];
  const float* stim  = (const float*)d_in[2];
  const float* W     = (const float*)d_in[3];
  const int T = in_sizes[0];
  const int N = in_sizes[2];

  // workspace layout: bf16 W | 4 bf16 R vectors | barrier counter
  unsigned short* wsW = (unsigned short*)d_ws;
  unsigned short* wsR = wsW + (size_t)N * N;
  unsigned*       bar = (unsigned*)(wsR + 4 * (size_t)N);

  hipMemsetAsync(bar, 0, 4 * sizeof(unsigned), stream);   // graph-capturable
  neuralode_rk4_persistent<<<NWG, BLOCK, 0, stream>>>(
      tg, state, stim, W, (float*)d_out, wsW, wsR, bar, N, T);
}
